// BertSelfAttention_86835648790780
// MI455X (gfx1250) — compile-verified, matrix operand order checked by reading, not run on hardware
//
#include <hip/hip_runtime.h>

// ---------------------------------------------------------------------------
// Problem constants (BertSelfAttention, relative_key_query)
// ---------------------------------------------------------------------------
#define BB     2
#define SS     2048
#define HH     16
#define DD     64
#define HIDD   1024
#define NDIST  4095          // 2*MAXPOS-1
#define NDISTP 4096          // padded (last row zero) so band reads stay in range

typedef _Float16 h16;
typedef __attribute__((ext_vector_type(16))) _Float16 v16h;
typedef __attribute__((ext_vector_type(8)))  _Float16 v8h;
typedef __attribute__((ext_vector_type(8)))  float    v8f;

__device__ __forceinline__ int lane_id() { return (int)(threadIdx.x & 31u); }

// ---------------------------------------------------------------------------
// Async global -> LDS copy (gfx1250 ASYNCcnt engine): each lane moves 16B
// from its global address to its LDS address.  Drained via s_wait_asynccnt.
// ---------------------------------------------------------------------------
__device__ __forceinline__ void async_copy_b128(unsigned lds_addr,
                                                const _Float16* g) {
  asm volatile("global_load_async_to_lds_b128 %0, %1, off"
               :: "v"(lds_addr), "v"(g)
               : "memory");
}
__device__ __forceinline__ void wait_async_le8() {
  asm volatile("s_wait_asynccnt 0x8" ::: "memory");
}
__device__ __forceinline__ void wait_async_0() {
  asm volatile("s_wait_asynccnt 0x0" ::: "memory");
}

// ---------------------------------------------------------------------------
// Load a 16x32 fp16 WMMA fragment (A-layout MxK == B-layout NxK under the
// CDNA5 striping: lanes 0-15 carry rows 0-15 with K 0-7 / 16-23 in VGPR 0-3 /
// 4-7; lanes 16-31 carry the same rows with K 8-15 / 24-31).  Source is a
// row-major [rows][stride] half array (global or LDS); two 16-byte loads.
// ---------------------------------------------------------------------------
__device__ __forceinline__ v16h load_frag16x32(const _Float16* p, int row_base,
                                               int stride, int k_off) {
  const int l  = lane_id();
  const int r  = row_base + (l & 15);
  const int kb = ((l >> 4) << 3) + k_off;    // 0 or 8, plus K-chunk offset
  const _Float16* q = p + (size_t)r * stride + kb;
  v8h lo = *(const v8h*)(q);                 // K = kb .. kb+7
  v8h hi = *(const v8h*)(q + 16);            // K = kb+16 .. kb+23
  return __builtin_shufflevector(lo, hi, 0, 1, 2, 3, 4, 5, 6, 7,
                                         8, 9, 10, 11, 12, 13, 14, 15);
}

__device__ __forceinline__ v8f wmma16(v16h a, v16h b, v8f c) {
  // D = A(16x32 f16) * B(32x16 f16) + C(16x16 f32)
  return __builtin_amdgcn_wmma_f32_16x16x32_f16(false, a, false, b,
                                                (short)0, c, false, false);
}

__device__ __forceinline__ float hmax16(float v) {
  v = fmaxf(v, __shfl_xor(v, 1, 32));
  v = fmaxf(v, __shfl_xor(v, 2, 32));
  v = fmaxf(v, __shfl_xor(v, 4, 32));
  v = fmaxf(v, __shfl_xor(v, 8, 32));
  return v;
}
__device__ __forceinline__ float hsum16(float v) {
  v += __shfl_xor(v, 1, 32);
  v += __shfl_xor(v, 2, 32);
  v += __shfl_xor(v, 4, 32);
  v += __shfl_xor(v, 8, 32);
  return v;
}

// ---------------------------------------------------------------------------
// Kernel 1: dist_emb (f32 [4095][64]) -> f16 [4096][64], last row zeroed.
// ---------------------------------------------------------------------------
__global__ void cvt_dist_kernel(const float* __restrict__ de,
                                _Float16* __restrict__ eh) {
  int idx = blockIdx.x * 256 + threadIdx.x;
  if (idx < NDISTP * DD) {
    eh[idx] = (idx < NDIST * DD) ? (_Float16)de[idx] : (_Float16)0.0f;
  }
}

// ---------------------------------------------------------------------------
// Kernel 2: fused projection  out[b,h,s,d] (f16) = hs @ W^T + bias
//   GEMM M=B*S=4096, N=HID=1024, K=HID=1024.  128 threads = 4 waves,
//   64x64 output tile per workgroup, each wave owns a 16-row M slice.
// ---------------------------------------------------------------------------
__global__ __launch_bounds__(128)
void qkv_kernel(const float* __restrict__ hs, const float* __restrict__ W,
                const float* __restrict__ bias, _Float16* __restrict__ dst) {
  __shared__ __align__(16) _Float16 ldsA[64 * 40];   // stride 40 halfs: 16B rows, bank-spread
  __shared__ __align__(16) _Float16 ldsB[64 * 40];

  const int wave = threadIdx.x >> 5;
  const int lane = lane_id();
  const int nloc = lane & 15;
  const int hlf  = lane >> 4;
  const int M0   = blockIdx.x * 64;
  const int N0   = blockIdx.y * 64;

  v8f acc[4] = {};

  for (int k0 = 0; k0 < HIDD; k0 += 32) {
    __syncthreads();
    // stage A (activations) and B (weight rows) as f16, row-major [64][32]
    for (int idx = threadIdx.x; idx < 64 * 32; idx += 128) {
      int rr = idx >> 5, cc = idx & 31;
      ldsA[rr * 40 + cc] = (_Float16)hs[(size_t)(M0 + rr) * HIDD + k0 + cc];
      ldsB[rr * 40 + cc] = (_Float16)W[(size_t)(N0 + rr) * HIDD + k0 + cc];
    }
    __syncthreads();

    v16h a = load_frag16x32(ldsA, wave * 16, 40, 0);
#pragma unroll
    for (int t = 0; t < 4; ++t) {
      v16h bf = load_frag16x32(ldsB, t * 16, 40, 0);
      acc[t] = wmma16(a, bf, acc[t]);
    }
  }

  // epilogue: + bias, scatter into heads-major [B][H][S][D] f16
#pragma unroll
  for (int t = 0; t < 4; ++t) {
#pragma unroll
    for (int i = 0; i < 8; ++i) {
      int m   = i + hlf * 8;
      int row = M0 + wave * 16 + m;          // flat (b, s)
      int o   = N0 + t * 16 + nloc;          // output feature
      float v = acc[t][i] + bias[o];
      int bidx = row >> 11, sidx = row & (SS - 1);
      int hh = o >> 6, d = o & 63;
      dst[(((size_t)(bidx * HH + hh) * SS) + sidx) * DD + d] = (_Float16)v;
    }
  }
}

// ---------------------------------------------------------------------------
// Kernel 3: V transpose  [B,H,S,D] -> [B,H,D,S] (f16), 64x64 LDS tiles.
//   Loads land in LDS via the async engine (coalesced 16B lane copies),
//   stores are contiguous along S.  One-time cost ~16 MB of traffic.
// ---------------------------------------------------------------------------
__global__ __launch_bounds__(256)
void vtr_kernel(const _Float16* __restrict__ v, _Float16* __restrict__ vt) {
  __shared__ __align__(16) _Float16 t[64 * 72];      // 144B rows: 16B aligned
  const int bh = blockIdx.y;
  const int s0 = blockIdx.x * 64;
  const _Float16* src = v + ((size_t)bh * SS + s0) * DD;   // [64 s][64 d]
  _Float16* dst = vt + (size_t)bh * DD * SS + s0;          // [64 d][2048 s]

  // async stage: 64 rows x 128B = 512 x 16B chunks over 256 threads
#pragma unroll
  for (int c = 0; c < 2; ++c) {
    int chunk = c * 256 + threadIdx.x;
    int row = chunk >> 3, col = chunk & 7;             // col: 16B unit
    unsigned laddr = (unsigned)(size_t)(t + row * 72 + col * 8);
    async_copy_b128(laddr, src + (size_t)row * DD + col * 8);
  }
  wait_async_0();
  __syncthreads();

  for (int idx = threadIdx.x; idx < 64 * 64; idx += 256) {
    int d = idx >> 6, j = idx & 63;                    // consecutive j -> contiguous store
    dst[(size_t)d * SS + j] = t[j * 72 + d];
  }
}

// ---------------------------------------------------------------------------
// Kernel 4: flash attention with relative_key_query bias.
//   128 threads = 4 waves, fully independent (no barriers).  Wave w handles
//   query rows [L0+16w, L0+16w+16); R swept in steps of 32.
//   K tiles are double-buffered in LDS via async-to-LDS copies issued one
//   step ahead (s_wait_asynccnt 0x8 leaves only the in-flight prefetch).
//   Per 16x16 subtile: score = Q K^T (2 WMMA); strip GEMMs Sq = Q E_band^T,
//   Sk = K E_band^T over the 32-row band j0 = L0 - R0c + 2032 (4+4 WMMA);
//   shear gather score[m,n] += Sq[m][m-n+15] + Sk[n][m-n+15]; *1/8 + mask.
//   Online softmax via 16-lane shuffles; P@V B-fragments load straight from
//   the transposed V (L2-resident).
// ---------------------------------------------------------------------------
__global__ __launch_bounds__(128)
void attn_kernel(const _Float16* __restrict__ qh, const _Float16* __restrict__ kh,
                 const _Float16* __restrict__ vt, const _Float16* __restrict__ eh,
                 const float* __restrict__ am, float* __restrict__ out) {
  __shared__ __align__(16) _Float16 ldsK[4][2][32 * 72]; // per-wave dbl-buffered K tile
  __shared__ __align__(16) float    ldsSq[4][16 * 32];   // per-wave rel_q strip
  __shared__ __align__(16) float    ldsSk[4][16 * 32];   // per-wave rel_k strip
  __shared__ __align__(16) _Float16 ldsP[4][16 * 32];    // per-wave probs tile

  const int wave = threadIdx.x >> 5;
  const int lane = lane_id();
  const int nloc = lane & 15;
  const int hlf  = lane >> 4;

  const int b  = blockIdx.z;
  const int h  = blockIdx.y;
  const int L0 = blockIdx.x * 64 + wave * 16;

  const _Float16* qbase = qh + (size_t)(b * HH + h) * SS * DD;   // [S][64]
  const _Float16* kbase = kh + (size_t)(b * HH + h) * SS * DD;   // [S][64]
  const _Float16* vtb   = vt + (size_t)(b * HH + h) * DD * SS;   // [64][S]
  const float*    mrow  = am + (size_t)b * SS;

  // Q fragments for this wave's 16 rows (K = 64 -> two 16x32 fragments)
  v16h qf0 = load_frag16x32(qbase, L0, DD, 0);
  v16h qf1 = load_frag16x32(qbase, L0, DD, 32);

  v8f O0 = {}, O1 = {}, O2 = {}, O3 = {};
  float mrun[8], srun[8];
#pragma unroll
  for (int i = 0; i < 8; ++i) { mrun[i] = -3.0e38f; srun[i] = 0.0f; }

  float*    sqs = ldsSq[wave];
  float*    sks = ldsSk[wave];
  _Float16* ps  = ldsP[wave];

  // ---- async prefetch of a 32x64 K tile: 256 x 16B chunks, 8 ops/wave ----
  auto prefetch_k = [&](_Float16* dstLds, int R) {
#pragma unroll
    for (int c = 0; c < 8; ++c) {
      int chunk = c * 32 + lane;
      int row = chunk >> 3, col = chunk & 7;           // col: 16B unit
      unsigned laddr = (unsigned)(size_t)(dstLds + row * 72 + col * 8);
      async_copy_b128(laddr, kbase + (size_t)(R + row) * DD + col * 8);
    }
  };

  prefetch_k(ldsK[wave][0], 0);

  for (int R0 = 0; R0 < SS; R0 += 32) {
    _Float16* cur = ldsK[wave][(R0 >> 5) & 1];
    _Float16* nxt = ldsK[wave][((R0 >> 5) & 1) ^ 1];
    const int Rn = (R0 + 32 < SS) ? (R0 + 32) : 0;
    prefetch_k(nxt, Rn);      // issue next step's tile
    wait_async_le8();         // only the just-issued 8 may remain -> `cur` ready

    v8f ptile[2];
#pragma unroll
    for (int rc = 0; rc < 2; ++rc) {
      const int R0c = R0 + rc * 16;

      // K fragments from the async-staged LDS tile (serve both B-operand of
      // QK^T and A-operand of the rel_k strip GEMM: identical layout)
      v16h kf0 = load_frag16x32(cur, rc * 16, 72, 0);
      v16h kf1 = load_frag16x32(cur, rc * 16, 72, 32);

      // raw scores
      v8f sc = {};
      sc = wmma16(qf0, kf0, sc);
      sc = wmma16(qf1, kf1, sc);

      // embedding band for this diagonal: rows j0 .. j0+31 of dist_emb
      const int j0 = L0 - R0c + 2032;        // L0 - R0c + (MAXPOS-1) - 15 >= 0
      const _Float16* eb = eh + (size_t)j0 * DD;
      // speculative prefetch of next step's band (global_prefetch_b8)
      __builtin_prefetch(eh + (size_t)(j0 >= 32 ? j0 - 32 : 0) * DD, 0, 0);

      v16h e00 = load_frag16x32(eb, 0,  DD, 0);
      v16h e01 = load_frag16x32(eb, 0,  DD, 32);
      v16h e10 = load_frag16x32(eb, 16, DD, 0);
      v16h e11 = load_frag16x32(eb, 16, DD, 32);

      v8f sq0 = {}, sq1 = {}, sk0 = {}, sk1 = {};
      sq0 = wmma16(qf0, e00, sq0); sq0 = wmma16(qf1, e01, sq0);
      sq1 = wmma16(qf0, e10, sq1); sq1 = wmma16(qf1, e11, sq1);
      sk0 = wmma16(kf0, e00, sk0); sk0 = wmma16(kf1, e01, sk0);
      sk1 = wmma16(kf0, e10, sk1); sk1 = wmma16(kf1, e11, sk1);

      // spill strips (C layout -> row-major [16][32]) for the shear gather
#pragma unroll
      for (int i = 0; i < 8; ++i) {
        int m = i + hlf * 8;
        sqs[m * 32 + nloc]      = sq0[i];
        sqs[m * 32 + 16 + nloc] = sq1[i];
        sks[m * 32 + nloc]      = sk0[i];
        sks[m * 32 + 16 + nloc] = sk1[i];
      }

      // shear-gather + scale + additive mask
      float mval = mrow[R0c + nloc];
#pragma unroll
      for (int i = 0; i < 8; ++i) {
        int m  = i + hlf * 8;
        int jj = m - nloc + 15;              // in [0, 30]
        sc[i] = (sc[i] + sqs[m * 32 + jj] + sks[nloc * 32 + jj]) * 0.125f + mval;
      }
      ptile[rc] = sc;
    }

    // ---- online softmax over these 32 columns ----
    float mnew[8], corr[8];
#pragma unroll
    for (int i = 0; i < 8; ++i) {
      float v = fmaxf(ptile[0][i], ptile[1][i]);
      v = hmax16(v);
      mnew[i] = fmaxf(mrun[i], v);
      corr[i] = __expf(mrun[i] - mnew[i]);
      mrun[i] = mnew[i];
    }
#pragma unroll
    for (int i = 0; i < 8; ++i) {
      float p0 = __expf(ptile[0][i] - mnew[i]);
      float p1 = __expf(ptile[1][i] - mnew[i]);
      ptile[0][i] = p0; ptile[1][i] = p1;
      srun[i] = srun[i] * corr[i] + hsum16(p0 + p1);
      O0[i] *= corr[i]; O1[i] *= corr[i]; O2[i] *= corr[i]; O3[i] *= corr[i];
    }

    // ---- stage P as f16 row-major [16][32] and run P @ V ----
#pragma unroll
    for (int i = 0; i < 8; ++i) {
      int m = i + hlf * 8;
      ps[m * 32 + nloc]      = (_Float16)ptile[0][i];
      ps[m * 32 + 16 + nloc] = (_Float16)ptile[1][i];
    }
    v16h pf  = load_frag16x32(ps, 0, 32, 0);
    // V^T B-fragments straight from global (L2-resident): rows = d, K = r
    v16h vf0 = load_frag16x32(vtb + R0, 0,  SS, 0);
    v16h vf1 = load_frag16x32(vtb + R0, 16, SS, 0);
    v16h vf2 = load_frag16x32(vtb + R0, 32, SS, 0);
    v16h vf3 = load_frag16x32(vtb + R0, 48, SS, 0);
    O0 = wmma16(pf, vf0, O0);
    O1 = wmma16(pf, vf1, O1);
    O2 = wmma16(pf, vf2, O2);
    O3 = wmma16(pf, vf3, O3);
  }

  // ---- epilogue: normalize and write ctx [B, S, H*D] f32 ----
#pragma unroll
  for (int i = 0; i < 8; ++i) {
    int m   = i + hlf * 8;
    int row = L0 + m;
    float inv = 1.0f / srun[i];
    float* orow = out + ((size_t)(b * SS + row)) * HIDD + h * DD;
    orow[0  + nloc] = O0[i] * inv;
    orow[16 + nloc] = O1[i] * inv;
    orow[32 + nloc] = O2[i] * inv;
    orow[48 + nloc] = O3[i] * inv;
  }
}

// ---------------------------------------------------------------------------
// Launcher
// ---------------------------------------------------------------------------
extern "C" void kernel_launch(void* const* d_in, const int* in_sizes, int n_in,
                              void* d_out, int out_size, void* d_ws, size_t ws_size,
                              hipStream_t stream) {
  const float* hs = (const float*)d_in[0];
  const float* am = (const float*)d_in[1];
  const float* Wq = (const float*)d_in[2];
  const float* bq = (const float*)d_in[3];
  const float* Wk = (const float*)d_in[4];
  const float* bk = (const float*)d_in[5];
  const float* Wv = (const float*)d_in[6];
  const float* bv = (const float*)d_in[7];
  const float* de = (const float*)d_in[8];

  const size_t NQKV = (size_t)BB * HH * SS * DD;     // 4,194,304 halfs each
  _Float16* qh  = (_Float16*)d_ws;
  _Float16* kh  = qh + NQKV;
  _Float16* vh  = kh + NQKV;
  _Float16* vth = vh + NQKV;                         // transposed V [B,H,D,S]
  _Float16* eh  = vth + NQKV;                        // [4096][64] f16

  cvt_dist_kernel<<<(NDISTP * DD + 255) / 256, 256, 0, stream>>>(de, eh);

  dim3 gproj(4096 / 64, HIDD / 64, 1);
  qkv_kernel<<<gproj, 128, 0, stream>>>(hs, Wq, bq, qh);
  qkv_kernel<<<gproj, 128, 0, stream>>>(hs, Wk, bk, kh);
  qkv_kernel<<<gproj, 128, 0, stream>>>(hs, Wv, bv, vh);

  vtr_kernel<<<dim3(SS / 64, BB * HH), 256, 0, stream>>>(vh, vth);

  attn_kernel<<<dim3(SS / 64, HH, BB), 128, 0, stream>>>(qh, kh, vth, eh, am,
                                                         (float*)d_out);
}
